// MaskedCausalAttention_3504693314105
// MI455X (gfx1250) — matrix operand order repaired and checked by reading, not compiled, not run on hardware
//
#include <hip/hip_runtime.h>
#include <hip/hip_bf16.h>

#define H 2048            // model dim
#define NHEADS 16
#define HD 128            // head dim
#define TSEQ 2048
#define BATCH 2
#define MROWS (BATCH*TSEQ)  // 4096

typedef __bf16 bf16;
typedef __attribute__((ext_vector_type(8)))  bf16  bf16x8;
typedef __attribute__((ext_vector_type(16))) bf16  bf16x16;
typedef __attribute__((ext_vector_type(8)))  float v8f;

// ---- Tensor Data Mover availability probe ----
#ifndef HAVE_TDM
#if defined(__has_builtin)
#if __has_builtin(__builtin_amdgcn_tensor_load_to_lds)
#define HAVE_TDM 1
#endif
#endif
#endif
#ifndef HAVE_TDM
#define HAVE_TDM 0
#endif

#if HAVE_TDM
typedef unsigned int uint32x4 __attribute__((ext_vector_type(4)));
typedef int          int32x4  __attribute__((ext_vector_type(4)));
typedef int          int32x8  __attribute__((ext_vector_type(8)));

// 2D tile DMA: global -> LDS. Units are 4-byte dwords (D#.data_size = 2).
// tile0_dw: tile width in dwords; tile1: rows; stride0_dw: global row stride.
__device__ __forceinline__ void tdm_load_2d(unsigned lds_off, const void* gptr,
                                            unsigned tile0_dw, unsigned tile1,
                                            unsigned stride0_dw) {
  unsigned long long ga = (unsigned long long)(size_t)gptr;
  uint32x4 g0;
  g0[0] = 1u;                                   // count=1, user descriptor
  g0[1] = lds_off;                              // lds_addr (bytes)
  g0[2] = (unsigned)(ga & 0xffffffffu);         // global_addr[31:0]
  g0[3] = (unsigned)((ga >> 32) & 0x01ffffffu)  // global_addr[56:32]
          | 0x80000000u;                        // type=2 (bits 127:126)
  const unsigned dim0 = 1u << 20;               // tensor dims large => no OOB clip
  const unsigned dim1 = 1u << 20;
  int32x8 g1;
  g1[0] = (int)(2u << 16);                      // data_size=2 (4B), no flags
  g1[1] = (int)((dim0 & 0xffffu) << 16);        // tensor_dim0[15:0]
  g1[2] = (int)(((dim0 >> 16) & 0xffffu) | ((dim1 & 0xffffu) << 16));
  g1[3] = (int)(((dim1 >> 16) & 0xffffu) | (tile0_dw << 16));   // tile_dim0
  g1[4] = (int)(tile1 & 0xffffu);               // tile_dim1 (tile_dim2=0)
  g1[5] = (int)stride0_dw;                      // tensor_dim0_stride[31:0]
  g1[6] = 0;                                    // stride0[47:32], stride1[15:0]
  g1[7] = 0;
  int32x4 z4 = {0, 0, 0, 0};
#if defined(__clang_major__) && (__clang_major__ >= 23)
  int32x8 z8 = {0, 0, 0, 0, 0, 0, 0, 0};
  __builtin_amdgcn_tensor_load_to_lds(g0, g1, z4, z4, z8, 0);
#else
  __builtin_amdgcn_tensor_load_to_lds(g0, g1, z4, z4, 0);
#endif
}
__device__ __forceinline__ unsigned lds_offset_of(const void* p) {
  return (unsigned)(size_t)p;  // flat LDS address: addr[31:0] is the LDS offset
}
#endif  // HAVE_TDM

__device__ __forceinline__ v8f wmma_bf16(bf16x16 a, bf16x16 b, v8f c) {
  return __builtin_amdgcn_wmma_f32_16x16x32_bf16(false, a, false, b, (short)0, c,
                                                 false, false);
}

__device__ __forceinline__ bf16x16 cat8(bf16x8 lo, bf16x8 hi) {
  return __builtin_shufflevector(lo, hi, 0,1,2,3,4,5,6,7,8,9,10,11,12,13,14,15);
}

// ---------------- fp32 -> bf16 conversion ----------------
__global__ void cvt_f32_bf16(const float* __restrict__ in, bf16* __restrict__ out,
                             int n) {
  int i4 = (blockIdx.x * blockDim.x + threadIdx.x) * 4;
  if (i4 < n) {
    float4 f = *(const float4*)(in + i4);
    out[i4 + 0] = (bf16)f.x;
    out[i4 + 1] = (bf16)f.y;
    out[i4 + 2] = (bf16)f.z;
    out[i4 + 3] = (bf16)f.w;
  }
}

// ---------------- bf16 GEMM: out = (A @ W + bias) * scale ----------------
// A: [M x H] bf16 row-major, W: [H x H] bf16 row-major.
// mode 0: write bf16 head-major [B][NH][T][HD]
// mode 1: write fp32 row-major [M][H] (final output)
#define BM 128
#define BN 128
#define BK 32

__global__ __launch_bounds__(256) void gemm_bf16(
    const bf16* __restrict__ A, const bf16* __restrict__ W,
    const float* __restrict__ bias, bf16* __restrict__ outHeads,
    float* __restrict__ outF32, float scale, int mode) {
  __shared__ __align__(64) bf16 As[2][BM * BK];   // [m][k], double-buffered
  __shared__ __align__(64) bf16 Bt[2][BN * BK];   // [n][k] (transposed on fill)

  const int tid  = threadIdx.x;
  const int lane = tid & 31;
  const int wave = tid >> 5;
  const int l15  = lane & 15;
  const int hi   = (lane >= 16) ? 1 : 0;
  const int mBase = blockIdx.y * BM;
  const int nBase = blockIdx.x * BN;

  v8f acc[8];
#pragma unroll
  for (int i = 0; i < 8; ++i) acc[i] = (v8f){0.f,0.f,0.f,0.f,0.f,0.f,0.f,0.f};

  const int rowA = wave * 16 + l15;
  const int khA  = hi ? 16 : 0;
  const int ksB  = hi ? 16 : 0;

#if HAVE_TDM
  const unsigned asOff = lds_offset_of(&As[0][0]);
#endif

  // W tile stage: hoist both global loads, then scatter-transpose into LDS.
  auto stageW = [&](int bb, int k0) {
    const int kk = tid >> 4;          // 0..15
    const int nc = (tid & 15) * 8;
    bf16x8 w0 = *(const bf16x8*)&W[(size_t)(k0 + kk) * H + nBase + nc];
    bf16x8 w1 = *(const bf16x8*)&W[(size_t)(k0 + kk + 16) * H + nBase + nc];
    if (k0 + 2 * BK < H)  // warm L2 two tiles ahead
      __builtin_prefetch(&W[(size_t)(k0 + 2 * BK + kk) * H + nBase + nc], 0, 3);
#pragma unroll
    for (int j = 0; j < 8; ++j) Bt[bb][(nc + j) * BK + kk] = w0[j];
#pragma unroll
    for (int j = 0; j < 8; ++j) Bt[bb][(nc + j) * BK + kk + 16] = w1[j];
  };
  auto stageA = [&](int bb, int k0) {
#if HAVE_TDM
    if (wave == 0)
      tdm_load_2d(asOff + (unsigned)bb * BM * BK * 2,
                  &A[(size_t)mBase * H + k0],
                  /*tile0_dw=*/BK / 2, /*tile1=*/BM, /*stride0_dw=*/H / 2);
#else
    const int r = tid >> 2;           // 0..63
    const int c = (tid & 3) * 8;
    bf16x8 a0 = *(const bf16x8*)&A[(size_t)(mBase + r) * H + k0 + c];
    bf16x8 a1 = *(const bf16x8*)&A[(size_t)(mBase + r + 64) * H + k0 + c];
    *(bf16x8*)&As[bb][r * BK + c] = a0;
    *(bf16x8*)&As[bb][(r + 64) * BK + c] = a1;
#endif
  };

  const int NK = H / BK;  // 64
  stageA(0, 0);
  stageW(0, 0);

  for (int ik = 0; ik < NK; ++ik) {
    const int buf = ik & 1;
#if HAVE_TDM
    if (wave == 0) __builtin_amdgcn_s_wait_tensorcnt(0);
#endif
    __syncthreads();  // stage(ik) complete & visible; readers of buf^1 done
    if (ik + 1 < NK) {
      stageA(buf ^ 1, (ik + 1) * BK);
      stageW(buf ^ 1, (ik + 1) * BK);
    }
    // compute on buf
    bf16x8 a0 = *(const bf16x8*)&As[buf][rowA * BK + khA];
    bf16x8 a1 = *(const bf16x8*)&As[buf][rowA * BK + 8 + khA];
    bf16x16 afrag = cat8(a0, a1);
#pragma unroll
    for (int nb = 0; nb < 8; ++nb) {
      bf16x16 bfrag = *(const bf16x16*)&Bt[buf][(nb * 16 + l15) * BK + ksB];
      acc[nb] = wmma_bf16(afrag, bfrag, acc[nb]);
    }
  }

  // store
#pragma unroll
  for (int nb = 0; nb < 8; ++nb) {
#pragma unroll
    for (int r = 0; r < 8; ++r) {
      int mrow = mBase + wave * 16 + r + hi * 8;
      int ncol = nBase + nb * 16 + l15;
      float v = (acc[nb][r] + bias[ncol]) * scale;
      if (mode == 0) {
        int b = mrow >> 11, t = mrow & (TSEQ - 1);
        int h = ncol >> 7,  d = ncol & (HD - 1);
        outHeads[(((size_t)b * NHEADS + h) * TSEQ + t) * HD + d] = (bf16)v;
      } else {
        outF32[(size_t)mrow * H + ncol] = v;
      }
    }
  }
}

// ---------------- flash attention (causal, online softmax) ----------------
#define QB 64   // q rows per workgroup (4 waves x 16)
#define SB 32   // keys per iteration (= WMMA K for PV)

__global__ __launch_bounds__(128) void attn_fwd(
    const bf16* __restrict__ Q, const bf16* __restrict__ K,
    const bf16* __restrict__ V, bf16* __restrict__ Ctx) {
  __shared__ __align__(64) bf16 Kt[2][SB * HD];    // [s][d]  32x128, dbl-buffered
  __shared__ __align__(64) bf16 Vt[2][HD * SB];    // [d][s] 128x32 (transposed)
  __shared__ __align__(64) bf16 Ps[4][16 * SB];    // per-wave P scratch 16x32

  const int tid  = threadIdx.x;
  const int lane = tid & 31;
  const int wave = tid >> 5;
  const int l15  = lane & 15;
  const int hi   = (lane >= 16) ? 1 : 0;

  const int qBase = blockIdx.x * QB;
  const int h = blockIdx.y;
  const int b = blockIdx.z;
  const size_t headOff = ((size_t)b * NHEADS + h) * TSEQ * HD;

  // resident Q fragments: this wave's 16 rows x HD
  const int qRow = qBase + wave * 16 + l15;
  bf16x16 qf[4];
  {
    const bf16* qp = Q + headOff + (size_t)qRow * HD;
    const int kh = hi ? 16 : 0;
#pragma unroll
    for (int dk = 0; dk < 4; ++dk) {
      bf16x8 a0 = *(const bf16x8*)&qp[dk * 32 + kh];
      bf16x8 a1 = *(const bf16x8*)&qp[dk * 32 + 8 + kh];
      qf[dk] = cat8(a0, a1);
    }
  }

  v8f o[8];
#pragma unroll
  for (int i = 0; i < 8; ++i) o[i] = (v8f){0.f,0.f,0.f,0.f,0.f,0.f,0.f,0.f};
  float mrow[8], lrow[8];
#pragma unroll
  for (int r = 0; r < 8; ++r) { mrow[r] = -INFINITY; lrow[r] = 0.f; }

#if HAVE_TDM
  const unsigned ktOff = lds_offset_of(&Kt[0][0]);
#endif

  // K tile: contiguous 8KB -> TDM; V tile: hoisted loads + transpose scatter.
  auto stageK = [&](int bb, int s0) {
#if HAVE_TDM
    if (wave == 0)
      tdm_load_2d(ktOff + (unsigned)bb * SB * HD * 2,
                  &K[headOff + (size_t)s0 * HD],
                  /*tile0_dw=*/(SB * HD) / 2, /*tile1=*/1,
                  /*stride0_dw=*/(SB * HD) / 2);
#else
    const int s_ = tid >> 4;          // 0..7
    const int dc = (tid & 15) * 8;
#pragma unroll
    for (int i = 0; i < 4; ++i)
      *(bf16x8*)&Kt[bb][(s_ + i * 8) * HD + dc] =
          *(const bf16x8*)&K[headOff + (size_t)(s0 + s_ + i * 8) * HD + dc];
#endif
  };
  auto stageV = [&](int bb, int s0) {
    const int s_ = tid >> 4;          // 0..7
    const int dc = (tid & 15) * 8;
    bf16x8 vv[4];
#pragma unroll
    for (int i = 0; i < 4; ++i)
      vv[i] = *(const bf16x8*)&V[headOff + (size_t)(s0 + s_ + i * 8) * HD + dc];
#pragma unroll
    for (int i = 0; i < 4; ++i)
#pragma unroll
      for (int j = 0; j < 8; ++j) Vt[bb][(dc + j) * SB + (s_ + i * 8)] = vv[i][j];
  };

  const int ksB = hi ? 16 : 0;
  const int nIter = (qBase + QB) / SB;
  stageK(0, 0);
  stageV(0, 0);

  for (int ik = 0; ik < nIter; ++ik) {
    const int buf = ik & 1;
    const int s0 = ik * SB;
#if HAVE_TDM
    if (wave == 0) __builtin_amdgcn_s_wait_tensorcnt(0);
#endif
    __syncthreads();  // stage(ik) complete; readers of buf^1 finished
    if (ik + 1 < nIter) {
      stageK(buf ^ 1, s0 + SB);
      stageV(buf ^ 1, s0 + SB);
    }

    // S = Q K^T : two 16x16 fragments covering 32 keys
    v8f sf[2];
#pragma unroll
    for (int nb = 0; nb < 2; ++nb) {
      v8f c = (v8f){0.f,0.f,0.f,0.f,0.f,0.f,0.f,0.f};
#pragma unroll
      for (int dk = 0; dk < 4; ++dk) {
        bf16x16 kf =
            *(const bf16x16*)&Kt[buf][(nb * 16 + l15) * HD + dk * 32 + ksB];
        c = wmma_bf16(qf[dk], kf, c);
      }
      sf[nb] = c;
    }

    // causal mask, per-row block max (16-lane halves hold one row each)
    float mblk[8];
#pragma unroll
    for (int r = 0; r < 8; ++r) {
      int row = qBase + wave * 16 + r + hi * 8;
#pragma unroll
      for (int nb = 0; nb < 2; ++nb) {
        int col = s0 + nb * 16 + l15;
        if (col > row) sf[nb][r] = -INFINITY;
      }
      float mx = fmaxf(sf[0][r], sf[1][r]);
      mx = fmaxf(mx, __shfl_xor(mx, 1, 16));
      mx = fmaxf(mx, __shfl_xor(mx, 2, 16));
      mx = fmaxf(mx, __shfl_xor(mx, 4, 16));
      mx = fmaxf(mx, __shfl_xor(mx, 8, 16));
      mblk[r] = mx;
    }

    // online softmax update + rescale accumulators
#pragma unroll
    for (int r = 0; r < 8; ++r) {
      float mnew  = fmaxf(mrow[r], mblk[r]);
      float alpha = __expf(mrow[r] - mnew);
      float psum = 0.f;
#pragma unroll
      for (int nb = 0; nb < 2; ++nb) {
        float p = __expf(sf[nb][r] - mnew);
        sf[nb][r] = p;
        psum += p;
      }
      psum += __shfl_xor(psum, 1, 16);
      psum += __shfl_xor(psum, 2, 16);
      psum += __shfl_xor(psum, 4, 16);
      psum += __shfl_xor(psum, 8, 16);
      lrow[r] = lrow[r] * alpha + psum;
      mrow[r] = mnew;
#pragma unroll
      for (int db = 0; db < 8; ++db) o[db][r] *= alpha;
    }

    // P (C-layout, fp32) -> bf16 -> LDS -> reload in A-fragment layout
#pragma unroll
    for (int r = 0; r < 8; ++r) {
      int prow = r + hi * 8;
      Ps[wave][prow * SB + l15]      = (bf16)sf[0][r];
      Ps[wave][prow * SB + 16 + l15] = (bf16)sf[1][r];
    }
    asm volatile("s_wait_dscnt 0" ::: "memory");
    bf16x16 pf;
    {
      const int kh = hi ? 16 : 0;
      bf16x8 p0 = *(const bf16x8*)&Ps[wave][l15 * SB + kh];
      bf16x8 p1 = *(const bf16x8*)&Ps[wave][l15 * SB + 8 + kh];
      pf = cat8(p0, p1);
    }

    // O += P @ V  (8 d-blocks of 16)
#pragma unroll
    for (int db = 0; db < 8; ++db) {
      bf16x16 vf = *(const bf16x16*)&Vt[buf][(db * 16 + l15) * SB + ksB];
      o[db] = wmma_bf16(pf, vf, o[db]);
    }
  }

  // normalize and store to [B][T][C] (col = h*128 + d)
#pragma unroll
  for (int db = 0; db < 8; ++db) {
#pragma unroll
    for (int r = 0; r < 8; ++r) {
      int row = qBase + wave * 16 + r + hi * 8;
      float v = o[db][r] / lrow[r];
      Ctx[((size_t)b * TSEQ + row) * H + h * HD + db * 16 + l15] = (bf16)v;
    }
  }
}

// ---------------- host launch ----------------
extern "C" void kernel_launch(void* const* d_in, const int* in_sizes, int n_in,
                              void* d_out, int out_size, void* d_ws, size_t ws_size,
                              hipStream_t stream) {
  const float* src = (const float*)d_in[0];
  const float* Wq  = (const float*)d_in[1];
  const float* bq  = (const float*)d_in[2];
  const float* Wk  = (const float*)d_in[3];
  const float* bk  = (const float*)d_in[4];
  const float* Wv  = (const float*)d_in[5];
  const float* bv  = (const float*)d_in[6];
  const float* Wo  = (const float*)d_in[7];
  const float* bo  = (const float*)d_in[8];
  float* out = (float*)d_out;

  char* ws = (char*)d_ws;
  size_t off = 0;
  const size_t srcN = (size_t)MROWS * H;       // 8M
  const size_t wN   = (size_t)H * H;           // 4M
  bf16* src_bf = (bf16*)(ws + off); off += srcN * 2;
  bf16* Wq_bf  = (bf16*)(ws + off); off += wN * 2;
  bf16* Wk_bf  = (bf16*)(ws + off); off += wN * 2;
  bf16* Wv_bf  = (bf16*)(ws + off); off += wN * 2;
  bf16* Wo_bf  = (bf16*)(ws + off); off += wN * 2;
  bf16* Q_bf   = (bf16*)(ws + off); off += srcN * 2;  // head-major [B][NH][T][HD]
  bf16* K_bf   = (bf16*)(ws + off); off += srcN * 2;
  bf16* V_bf   = (bf16*)(ws + off); off += srcN * 2;
  bf16* Ctx_bf = (bf16*)(ws + off); off += srcN * 2;  // [B*T][H]
  (void)ws_size; (void)in_sizes; (void)n_in; (void)out_size;

  // 1) convert fp32 -> bf16
  cvt_f32_bf16<<<(int)(srcN / 1024), 256, 0, stream>>>(src, src_bf, (int)srcN);
  cvt_f32_bf16<<<(int)(wN / 1024),   256, 0, stream>>>(Wq,  Wq_bf,  (int)wN);
  cvt_f32_bf16<<<(int)(wN / 1024),   256, 0, stream>>>(Wk,  Wk_bf,  (int)wN);
  cvt_f32_bf16<<<(int)(wN / 1024),   256, 0, stream>>>(Wv,  Wv_bf,  (int)wN);
  cvt_f32_bf16<<<(int)(wN / 1024),   256, 0, stream>>>(Wo,  Wo_bf,  (int)wN);

  // 2) projections (Q pre-scaled by 1/sqrt(HD))
  dim3 ggrid(H / BN, MROWS / BM);
  const float qscale = 0.08838834764831843f;  // 1/sqrt(128)
  gemm_bf16<<<ggrid, 256, 0, stream>>>(src_bf, Wq_bf, bq, Q_bf, nullptr, qscale, 0);
  gemm_bf16<<<ggrid, 256, 0, stream>>>(src_bf, Wk_bf, bk, K_bf, nullptr, 1.0f, 0);
  gemm_bf16<<<ggrid, 256, 0, stream>>>(src_bf, Wv_bf, bv, V_bf, nullptr, 1.0f, 0);

  // 3) causal flash attention -> Ctx [B*T][H]
  dim3 agrid(TSEQ / QB, NHEADS, BATCH);
  attn_fwd<<<agrid, 128, 0, stream>>>(Q_bf, K_bf, V_bf, Ctx_bf);

  // 4) output projection -> fp32 d_out (+ bo)
  gemm_bf16<<<ggrid, 256, 0, stream>>>(Ctx_bf, Wo_bf, bo, nullptr, out, 1.0f, 1);
}
